// CausalTransformer_81269371175347
// MI455X (gfx1250) — matrix operand order repaired and checked by reading, not compiled, not run on hardware
//
#include <hip/hip_runtime.h>

// ---------------------------------------------------------------------------
// MADE causal-attention net on MI455X (gfx1250), bf16 WMMA everywhere.
// Layouts: activations position-major  [P = 512*128, channels], channels padded
// to a multiple of 32 (bf16 WMMA K step). Weights pre-converted per call to
// zero-padded bf16 with the MADE masks folded in.
// GEMM inner loop: double-buffered LDS tiles filled with
// global_load_async_to_lds_b128 (ASYNCcnt) for full tiles, WMMA 16x16x32 bf16.
// ---------------------------------------------------------------------------

typedef __attribute__((ext_vector_type(16))) __bf16 v16bf;
typedef __attribute__((ext_vector_type(8)))  __bf16 v8bf;
typedef __attribute__((ext_vector_type(8)))  float  v8f;

#define NSMP 512
#define DPOS 128
#define PTOT (NSMP * DPOS)   // 65536 positions
#define LDSP 40              // LDS row pitch in bf16 (bank spread)
#define TILEB (128 * LDSP)   // one LDS tile in bf16 elements

// ---------------------------- workspace layout -----------------------------
static constexpr size_t AL256(size_t x) { return (x + 255) & ~(size_t)255; }

static constexpr size_t OFF_E  = 0;                                              // bf16 [P,544] staging (also Abf for final)
static constexpr size_t OFF_H1 = AL256(OFF_E  + (size_t)PTOT * 544 * 2);         // f32 [P,258] (also inputs_bf at start)
static constexpr size_t OFF_G2 = AL256(OFF_H1 + (size_t)PTOT * 258 * 4);         // f32 [P,258]
static constexpr size_t OFF_X  = AL256(OFF_G2 + (size_t)PTOT * 258 * 4);         // f32 [P,256] hidden state
static constexpr size_t OFF_KB = AL256(OFF_X  + (size_t)PTOT * 256 * 4);         // bf16 [P,32] keys (pad 16->32)
static constexpr size_t OFF_QB = AL256(OFF_KB + (size_t)PTOT * 32 * 2);          // bf16 [P,32] queries
static constexpr size_t OFF_VB = AL256(OFF_QB + (size_t)PTOT * 32 * 2);          // bf16 [P,128] values
static constexpr size_t OFF_VT = AL256(OFF_VB + (size_t)PTOT * 128 * 2);         // bf16 [n,128vch,128pos] V transposed
static constexpr size_t OFF_S  = AL256(OFF_VT + (size_t)NSMP * 128 * 128 * 2);   // f32  [n,128,128] scores
static constexpr size_t OFF_SB = AL256(OFF_S  + (size_t)NSMP * 128 * 128 * 4);   // bf16 [n,128,128] softmax
static constexpr size_t OFF_AV = AL256(OFF_SB + (size_t)NSMP * 128 * 128 * 2);   // f32  [P,128] attn output
static constexpr size_t OFF_WI = AL256(OFF_AV + (size_t)PTOT * 128 * 4);         // bf16 [128pos][256][160] init W (masked)
static constexpr size_t OFF_WF = AL256(OFF_WI + (size_t)128 * 256 * 160 * 2);    // bf16 [2064,32768] final W (masked)
static constexpr size_t OFF_WB = AL256(OFF_WF + (size_t)2064 * 32768 * 2);       // per-block bf16 weights

// per-block weight sub-offsets (bf16 elements)
static constexpr size_t BO_K_IN   = 0;
static constexpr size_t BO_K_OUT  = BO_K_IN   + (size_t)258 * 544;
static constexpr size_t BO_Q_IN   = BO_K_OUT  + (size_t)516 * 544;
static constexpr size_t BO_Q_OUT  = BO_Q_IN   + (size_t)257 * 544;
static constexpr size_t BO_V_IN   = BO_Q_OUT  + (size_t)514 * 544;
static constexpr size_t BO_V_OUT  = BO_V_IN   + (size_t)258 * 544;
static constexpr size_t BO_NK     = BO_V_OUT  + (size_t)516 * 544;
static constexpr size_t BO_NQ     = BO_NK     + (size_t)16 * 288;
static constexpr size_t BO_NV     = BO_NQ     + (size_t)16 * 288;
static constexpr size_t BO_O_IN   = BO_NV     + (size_t)128 * 288;
static constexpr size_t BO_O_OUT  = BO_O_IN   + (size_t)256 * 512;
static constexpr size_t BO_O_SKIP = BO_O_OUT  + (size_t)512 * 512;
static constexpr size_t BLK_ELEMS = BO_O_SKIP + (size_t)256 * 256;
static constexpr size_t SZ_WB     = AL256(BLK_ELEMS * 2);

// ------------------------------- GEMM kernel -------------------------------
// C[M,N](+bias) = A[M,Kpad] @ W[N,Kpad]^T, batched via blockIdx.z.
// flags bit0: store bf16 (zero-padded to Nstore); bit1: accumulate into f32 C.
__global__ __launch_bounds__(256)
void gemm_kernel(const __bf16* __restrict__ A, int lda, long long sA,
                 const __bf16* __restrict__ W, int ldw, long long sW,
                 const float* __restrict__ bias, int bstride, long long bzs,
                 void* __restrict__ Cv, int ldc, long long sC,
                 int M, int N, int Kpad, int Nstore, int flags)
{
    __shared__ __bf16 lA[2 * TILEB];
    __shared__ __bf16 lW[2 * TILEB];

    const int tid  = threadIdx.x;
    const int z    = blockIdx.z;
    A += (long long)z * sA;
    W += (long long)z * sW;
    const int bm   = blockIdx.y * 128;
    const int bn   = blockIdx.x * 128;
    const int w    = tid >> 5;          // wave id (wave32)
    const int lane = tid & 31;
    const int wr   = w & 3;             // 4 wave-rows x 2 wave-cols
    const int wc   = w >> 2;
    const int lrow = lane & 15;
    const int hi   = lane >> 4;

    v8f acc[2][4] = {};
    const int nK = Kpad >> 5;

    // WMMA compute from LDS buffer bsel (contiguous k order, same in A and B)
    auto compute = [&](int bsel) {
        const int ba = bsel * TILEB;
        union U { v16bf v; v8bf h[2]; };
        U a[2], b[4];
        const int kbA = hi ? 16 : 0;  // A: lanes 0-15 K0..15, lanes 16-31 K16..31
#pragma unroll
        for (int i = 0; i < 2; ++i) {
            int row = wr * 32 + i * 16 + lrow;
            a[i].h[0] = *(const v8bf*)&lA[ba + row * LDSP + kbA];
            a[i].h[1] = *(const v8bf*)&lA[ba + row * LDSP + kbA + 8];
        }
        const int kbW = hi ? 16 : 0;  // B: lanes 0-15 K0..15, lanes 16-31 K16..31
#pragma unroll
        for (int j = 0; j < 4; ++j) {
            int col = wc * 64 + j * 16 + lrow;
            b[j].h[0] = *(const v8bf*)&lW[ba + col * LDSP + kbW];
            b[j].h[1] = *(const v8bf*)&lW[ba + col * LDSP + kbW + 8];
        }
#pragma unroll
        for (int i = 0; i < 2; ++i)
#pragma unroll
            for (int j = 0; j < 4; ++j)
                acc[i][j] = __builtin_amdgcn_wmma_f32_16x16x32_bf16(
                    false, a[i].v, false, b[j].v, (short)0, acc[i][j], false, false);
    };

    const bool full = (bm + 128 <= M) && (bn + 128 <= N);
    if (full) {
        // ---- async double-buffered pipeline (ASYNCcnt) ----
        auto issue = [&](int kk, int bsel) {
            const int ba = bsel * TILEB;
            const int k0 = kk << 5;
#pragma unroll
            for (int t = 0; t < 2; ++t) {
                int c  = tid + (t << 8);     // 512 chunks of 8 bf16
                int r  = c >> 2;
                int c8 = (c & 3) << 3;
                const __bf16* ga = A + (size_t)(bm + r) * lda + k0 + c8;
                unsigned la = (unsigned)(size_t)&lA[ba + r * LDSP + c8]; // LDS byte offset (addr[31:0])
                asm volatile("global_load_async_to_lds_b128 %0, %1, off"
                             :: "v"(la), "v"(ga) : "memory");
                const __bf16* gw = W + (size_t)(bn + r) * ldw + k0 + c8;
                unsigned lw = (unsigned)(size_t)&lW[ba + r * LDSP + c8];
                asm volatile("global_load_async_to_lds_b128 %0, %1, off"
                             :: "v"(lw), "v"(gw) : "memory");
            }
        };
        issue(0, 0);
        for (int kk = 0; kk < nK; ++kk) {
            asm volatile("s_wait_asynccnt 0x0" ::: "memory");
            __syncthreads();                       // tile kk resident for all waves
            if (kk + 1 < nK) issue(kk + 1, (kk + 1) & 1);  // prefetch next tile async
            compute(kk & 1);
            __syncthreads();                       // done reading tile kk
        }
    } else {
        // ---- edge tiles: synchronous zero-filled loads (buffer 0) ----
        for (int kk = 0; kk < nK; ++kk) {
            const int k0 = kk << 5;
#pragma unroll
            for (int t = 0; t < 2; ++t) {
                int c  = tid + (t << 8);
                int r  = c >> 2;
                int c8 = (c & 3) << 3;
                v8bf av = {};
                int gr = bm + r;
                if (gr < M) {
                    av = *(const v8bf*)(A + (size_t)gr * lda + k0 + c8);
                    if (kk + 1 < nK)
                        __builtin_prefetch((const void*)(A + (size_t)gr * lda + k0 + 32 + c8), 0, 1);
                }
                *(v8bf*)&lA[r * LDSP + c8] = av;
                v8bf wv = {};
                int gn = bn + r;
                if (gn < N) {
                    wv = *(const v8bf*)(W + (size_t)gn * ldw + k0 + c8);
                    if (kk + 1 < nK)
                        __builtin_prefetch((const void*)(W + (size_t)gn * ldw + k0 + 32 + c8), 0, 1);
                }
                *(v8bf*)&lW[r * LDSP + c8] = wv;
            }
            __syncthreads();
            compute(0);
            __syncthreads();
        }
    }

    const bool obf  = (flags & 1) != 0;
    const bool accm = (flags & 2) != 0;
    float*  Cf = (float*)Cv;
    __bf16* Cb = (__bf16*)Cv;
    const size_t cbase = (size_t)((long long)z * sC);
#pragma unroll
    for (int i = 0; i < 2; ++i) {
#pragma unroll
        for (int j = 0; j < 4; ++j) {
            const int gc = bn + wc * 64 + j * 16 + lrow;
#pragma unroll
            for (int e = 0; e < 8; ++e) {
                const int gr = bm + wr * 32 + i * 16 + e + hi * 8;
                if (gr >= M) continue;
                float v = acc[i][j][e];
                if (bias && gc < N) v += bias[(size_t)z * bzs + (size_t)gc * bstride];
                const size_t off = cbase + (size_t)gr * ldc + gc;
                if (obf) {
                    if (gc < Nstore) Cb[off] = (__bf16)((gc < N) ? v : 0.f);
                } else if (gc < N) {
                    if (accm) Cf[off] += v; else Cf[off] = v;
                }
            }
        }
    }
}

// --------------------------- elementwise kernels ---------------------------
__device__ __forceinline__ float sigmoidf_(float x) { return 1.f / (1.f + expf(-x)); }
__device__ __forceinline__ float siluf_(float x)    { return x / (1.f + expf(-x)); }
__device__ __forceinline__ float bgval_(int pos)    { return ((float)pos - 64.f) / 128.f; }

// virtual input: mode 0 = plain [P,xld]; 1 = IXB [inp, X, bg]; 2 = XB [X, bg]
__device__ __forceinline__ float xin_get(int mode, const float* __restrict__ xbuf, int xld,
                                         const float* __restrict__ inputs, int p, int c)
{
    if (mode == 1) {
        if (c == 0)   return inputs[(size_t)(p >> 7) * 129 + 1 + (p & 127)];
        if (c <= 256) return xbuf[(size_t)p * 256 + (c - 1)];
        return bgval_(p & 127);
    }
    if (mode == 2) {
        if (c < 256) return xbuf[(size_t)p * 256 + c];
        return bgval_(p & 127);
    }
    return xbuf[(size_t)p * xld + c];
}

// out[P,wout] bf16 = concat_elu(xin)[silu(x), silu(-x)], zero padded
__global__ __launch_bounds__(256)
void celu_kernel(int mode, const float* __restrict__ xbuf, int xld,
                 const float* __restrict__ inputs, int f, int wout,
                 __bf16* __restrict__ out)
{
    size_t idx = (size_t)blockIdx.x * 256 + threadIdx.x;
    size_t total = (size_t)PTOT * wout;
    if (idx >= total) return;
    int c = (int)(idx % wout);
    int p = (int)(idx / wout);
    float r = 0.f;
    if (c < f)           r = siluf_(xin_get(mode, xbuf, xld, inputs, p, c));
    else if (c < 2 * f)  r = siluf_(-xin_get(mode, xbuf, xld, inputs, p, c - f));
    out[idx] = (__bf16)r;
}

// res = xin + c0 * sigmoid(c1); outmode 0: f32 [P,ldo]; 1: bf16 [P,wpad] zero padded
__global__ __launch_bounds__(256)
void residual_kernel(int mode, const float* __restrict__ xbuf, int xld,
                     const float* __restrict__ inputs,
                     const float* __restrict__ c0, int ld0,
                     const float* __restrict__ c1, int ld1,
                     int f, int outmode, float* __restrict__ outf, int ldo,
                     __bf16* __restrict__ outb, int wpad)
{
    int w = outmode ? wpad : f;
    size_t idx = (size_t)blockIdx.x * 256 + threadIdx.x;
    size_t total = (size_t)PTOT * w;
    if (idx >= total) return;
    int c = (int)(idx % w);
    int p = (int)(idx / w);
    if (outmode) {
        float r = 0.f;
        if (c < f)
            r = xin_get(mode, xbuf, xld, inputs, p, c)
              + c0[(size_t)p * ld0 + c] * sigmoidf_(c1[(size_t)p * ld1 + c]);
        outb[(size_t)p * wpad + c] = (__bf16)r;
    } else {
        outf[(size_t)p * ldo + c] = xin_get(mode, xbuf, xld, inputs, p, c)
              + c0[(size_t)p * ld0 + c] * sigmoidf_(c1[(size_t)p * ld1 + c]);
    }
}

// causal zero-masked renormalized softmax; one wave per row; writes bf16
__global__ __launch_bounds__(256)
void softmax_kernel(const float* __restrict__ S, __bf16* __restrict__ Sb)
{
    int row  = blockIdx.x * 8 + (threadIdx.x >> 5);
    int lane = threadIdx.x & 31;
    int q    = row & 127;
    const float* r = S + (size_t)row * 128;
    float v[4];
    float mx = -3.0e38f;
#pragma unroll
    for (int t = 0; t < 4; ++t) {
        int k = lane + 32 * t;
        v[t] = (k < q) ? r[k] : 0.f;   // zero masking as in reference
        mx = fmaxf(mx, v[t]);
    }
#pragma unroll
    for (int off = 16; off >= 1; off >>= 1) mx = fmaxf(mx, __shfl_xor(mx, off, 32));
    float e[4], sAll = 0.f, sLt = 0.f;
#pragma unroll
    for (int t = 0; t < 4; ++t) {
        int k = lane + 32 * t;
        e[t] = expf(v[t] - mx);
        sAll += e[t];
        if (k < q) sLt += e[t];
    }
#pragma unroll
    for (int off = 16; off >= 1; off >>= 1) {
        sAll += __shfl_xor(sAll, off, 32);
        sLt  += __shfl_xor(sLt,  off, 32);
    }
    // s = (e/Z) / (sLt/Z + eps) = e / (sLt + eps*Z)
    float denom = sLt + 1e-7f * sAll;
#pragma unroll
    for (int t = 0; t < 4; ++t) {
        int k = lane + 32 * t;
        Sb[(size_t)row * 128 + k] = (__bf16)((k < q) ? e[t] / denom : 0.f);
    }
}

// ------------------------------ prep kernels -------------------------------
__global__ __launch_bounds__(256)
void prep_pad_kernel(const float* __restrict__ w, __bf16* __restrict__ o,
                     int N, int K, int Kpad)
{
    size_t idx = (size_t)blockIdx.x * 256 + threadIdx.x;
    size_t total = (size_t)N * Kpad;
    if (idx >= total) return;
    int k = (int)(idx % Kpad), n = (int)(idx / Kpad);
    o[idx] = (k < K) ? (__bf16)w[(size_t)n * K + k] : (__bf16)0.f;
}

__global__ __launch_bounds__(256)
void prep_init_kernel(const float* __restrict__ w, __bf16* __restrict__ o)
{   // o[pos][h][i<160]; MADE input mask: keep i <= pos
    size_t idx = (size_t)blockIdx.x * 256 + threadIdx.x;
    size_t total = (size_t)128 * 256 * 160;
    if (idx >= total) return;
    int i   = (int)(idx % 160);
    int h   = (int)((idx / 160) % 256);
    int pos = (int)(idx / (160 * 256));
    float v = 0.f;
    if (i <= pos) v = w[((size_t)h * 128 + pos) * 129 + i];
    o[idx] = (__bf16)v;
}

__global__ __launch_bounds__(256)
void prep_final_kernel(const float* __restrict__ w, __bf16* __restrict__ o)
{   // MADE output mask: keep (o_row/16) > (j % 128)
    size_t idx = (size_t)blockIdx.x * 256 + threadIdx.x;
    size_t total = (size_t)2064 * 32768;
    if (idx >= total) return;
    int j = (int)(idx & 32767), r = (int)(idx >> 15);
    o[idx] = ((r >> 4) > (j & 127)) ? (__bf16)w[idx] : (__bf16)0.f;
}

__global__ __launch_bounds__(256)
void inputs_bf_kernel(const float* __restrict__ in, __bf16* __restrict__ o)
{
    size_t idx = (size_t)blockIdx.x * 256 + threadIdx.x;
    size_t total = (size_t)NSMP * 160;
    if (idx >= total) return;
    int i = (int)(idx % 160), n = (int)(idx / 160);
    o[idx] = (i < 129) ? (__bf16)in[(size_t)n * 129 + i] : (__bf16)0.f;
}

__global__ __launch_bounds__(256)
void relu_abf_kernel(const float* __restrict__ X, __bf16* __restrict__ o)
{   // Abf[n, h*128+pos] = relu(X[(n*128+pos), h])
    size_t idx = (size_t)blockIdx.x * 256 + threadIdx.x;
    size_t total = (size_t)NSMP * 32768;
    if (idx >= total) return;
    int j = (int)(idx & 32767), n = (int)(idx >> 15);
    int h = j >> 7, pos = j & 127;
    float v = X[((size_t)n * 128 + pos) * 256 + h];
    o[idx] = (__bf16)(v > 0.f ? v : 0.f);
}

__global__ __launch_bounds__(256)
void vt_kernel(const __bf16* __restrict__ VB, __bf16* __restrict__ VT)
{   // VT[n][vch][j] = VB[(n*128+j)][vch]
    size_t idx = (size_t)blockIdx.x * 256 + threadIdx.x;
    size_t total = (size_t)NSMP * 128 * 128;
    if (idx >= total) return;
    int j   = (int)(idx & 127);
    int vch = (int)((idx >> 7) & 127);
    int n   = (int)(idx >> 14);
    VT[idx] = VB[(((size_t)n * 128 + j) << 7) + vch];
}

// ------------------------------- host helpers ------------------------------
static inline unsigned g1(size_t total) { return (unsigned)((total + 255) / 256); }

static void gemm_go(hipStream_t s, const __bf16* A, int lda, long long sA,
                    const __bf16* W, int ldw, long long sW,
                    const float* bias, int bstride, long long bzs,
                    void* C, int ldc, long long sC,
                    int M, int N, int Kpad, int Nstore, int flags, int batch)
{
    int nmax = N > Nstore ? N : Nstore;
    dim3 grd((nmax + 127) / 128, (M + 127) / 128, batch);
    gemm_kernel<<<grd, 256, 0, s>>>(A, lda, sA, W, ldw, sW, bias, bstride, bzs,
                                    C, ldc, sC, M, N, Kpad, Nstore, flags);
}

struct WPrep { int rel; size_t off; int N, K, Kp; };
static const WPrep g_wps[12] = {
    { 0, BO_K_IN,   258, 516, 544}, { 2, BO_K_OUT,  516, 516, 544},
    { 4, BO_Q_IN,   257, 514, 544}, { 6, BO_Q_OUT,  514, 514, 544},
    { 8, BO_V_IN,   258, 516, 544}, {10, BO_V_OUT,  516, 516, 544},
    {12, BO_NK,      16, 258, 288}, {14, BO_NQ,      16, 257, 288},
    {16, BO_NV,     128, 258, 288}, {18, BO_O_IN,   256, 512, 512},
    {20, BO_O_OUT,  512, 512, 512}, {22, BO_O_SKIP, 256, 256, 256},
};

static void run_block(hipStream_t s, void* const* d_in, char* ws, int b)
{
    const float* inputs = (const float*)d_in[0];
    const int ib = 5 + 24 * b;
    __bf16* WB = (__bf16*)(ws + OFF_WB + (size_t)b * SZ_WB);
    __bf16* E  = (__bf16*)(ws + OFF_E);
    float*  H1 = (float*)(ws + OFF_H1);
    float*  G2 = (float*)(ws + OFF_G2);
    float*  X  = (float*)(ws + OFF_X);
    __bf16* KB = (__bf16*)(ws + OFF_KB);
    __bf16* QB = (__bf16*)(ws + OFF_QB);
    __bf16* VB = (__bf16*)(ws + OFF_VB);
    __bf16* VT = (__bf16*)(ws + OFF_VT);
    float*  S  = (float*)(ws + OFF_S);
    __bf16* SB = (__bf16*)(ws + OFF_SB);
    float*  AV = (float*)(ws + OFF_AV);
    auto F = [&](int i) { return (const float*)d_in[i]; };

    // one GRN (k/q/v) path: res_attn -> nin -> bf16 projection out
    struct Path { int src; int f; size_t win, wout, nin; int bi_in, bi_out, bi_n;
                  int nN; __bf16* proj; int projPad; };
    Path paths[3] = {
        {1, 258, BO_K_IN, BO_K_OUT, BO_NK, ib + 1,  ib + 3,  ib + 13, 16,  KB, 32},
        {2, 257, BO_Q_IN, BO_Q_OUT, BO_NQ, ib + 5,  ib + 7,  ib + 15, 16,  QB, 32},
        {1, 258, BO_V_IN, BO_V_OUT, BO_NV, ib + 9,  ib + 11, ib + 17, 128, VB, 128},
    };
    for (int t = 0; t < 3; ++t) {
        const Path& P = paths[t];
        // E = concat_elu(xin)   [P, 544]
        celu_kernel<<<g1((size_t)PTOT * 544), 256, 0, s>>>(P.src, X, 256, inputs, P.f, 544, E);
        // H1 = E @ W_in^T + b_in   [P, f] f32
        gemm_go(s, E, 544, 0, WB + P.win, 544, 0, F(P.bi_in), 1, 0,
                H1, 258, 0, PTOT, P.f, 544, P.f, 0, 1);
        // E = concat_elu(H1)
        celu_kernel<<<g1((size_t)PTOT * 544), 256, 0, s>>>(0, H1, 258, inputs, P.f, 544, E);
        // c0 -> H1, c1 -> G2
        gemm_go(s, E, 544, 0, WB + P.wout, 544, 0, F(P.bi_out), 1, 0,
                H1, 258, 0, PTOT, P.f, 544, P.f, 0, 1);
        gemm_go(s, E, 544, 0, WB + P.wout + (size_t)P.f * 544, 544, 0, F(P.bi_out) + P.f, 1, 0,
                G2, 258, 0, PTOT, P.f, 544, P.f, 0, 1);
        // E = bf16(xin + c0*sigmoid(c1))  [P, 288] zero padded
        residual_kernel<<<g1((size_t)PTOT * 288), 256, 0, s>>>(P.src, X, 256, inputs,
                H1, 258, G2, 258, P.f, 1, nullptr, 0, E, 288);
        // proj = E @ nin^T + b  -> bf16, zero padded to projPad
        gemm_go(s, E, 288, 0, WB + P.nin, 288, 0, F(P.bi_n), 1, 0,
                P.proj, P.projPad, 0, PTOT, P.nN, 288, P.projPad, 1, 1);
    }

    // scores S[n][q][k] = Q . K  (batched, K padded to 32)
    gemm_go(s, QB, 32, (long long)128 * 32, KB, 32, (long long)128 * 32,
            nullptr, 0, 0, S, 128, (long long)128 * 128,
            128, 128, 32, 128, 0, NSMP);
    softmax_kernel<<<PTOT / 8, 256, 0, s>>>(S, SB);
    vt_kernel<<<g1((size_t)NSMP * 128 * 128), 256, 0, s>>>(VB, VT);
    // AV[n][q][vch] = sum_j SB[q][j] * VT[vch][j]  (batched)
    gemm_go(s, SB, 128, (long long)128 * 128, VT, 128, (long long)128 * 128,
            nullptr, 0, 0, AV, 128, (long long)128 * 128,
            128, 128, 128, 128, 0, NSMP);

    // grn_out with skip from AV, residual into X in place
    celu_kernel<<<g1((size_t)PTOT * 512), 256, 0, s>>>(0, X, 256, inputs, 256, 512, E);
    gemm_go(s, E, 512, 0, WB + BO_O_IN, 512, 0, F(ib + 19), 1, 0,
            H1, 258, 0, PTOT, 256, 512, 256, 0, 1);
    celu_kernel<<<g1((size_t)PTOT * 256), 256, 0, s>>>(0, AV, 128, inputs, 128, 256, E);
    gemm_go(s, E, 256, 0, WB + BO_O_SKIP, 256, 0, F(ib + 23), 1, 0,
            H1, 258, 0, PTOT, 256, 256, 256, 2 /*acc*/, 1);
    celu_kernel<<<g1((size_t)PTOT * 512), 256, 0, s>>>(0, H1, 258, inputs, 256, 512, E);
    gemm_go(s, E, 512, 0, WB + BO_O_OUT, 512, 0, F(ib + 21), 1, 0,
            H1, 258, 0, PTOT, 256, 512, 256, 0, 1);
    gemm_go(s, E, 512, 0, WB + BO_O_OUT + (size_t)256 * 512, 512, 0, F(ib + 21) + 256, 1, 0,
            G2, 258, 0, PTOT, 256, 512, 256, 0, 1);
    residual_kernel<<<g1((size_t)PTOT * 256), 256, 0, s>>>(0, X, 256, inputs,
            H1, 258, G2, 258, 256, 0, X, 256, nullptr, 0);
}

// --------------------------------- entry ----------------------------------
extern "C" void kernel_launch(void* const* d_in, const int* in_sizes, int n_in,
                              void* d_out, int out_size, void* d_ws, size_t ws_size,
                              hipStream_t stream)
{
    (void)in_sizes; (void)n_in; (void)out_size; (void)ws_size;
    char* ws = (char*)d_ws;
    const float* inputs = (const float*)d_in[0];

    // --- weight prep (masked bf16, zero padded) ---
    prep_init_kernel<<<g1((size_t)128 * 256 * 160), 256, 0, stream>>>(
        (const float*)d_in[1], (__bf16*)(ws + OFF_WI));
    prep_final_kernel<<<g1((size_t)2064 * 32768), 256, 0, stream>>>(
        (const float*)d_in[3], (__bf16*)(ws + OFF_WF));
    for (int b = 0; b < 4; ++b) {
        const int ib = 5 + 24 * b;
        __bf16* WB = (__bf16*)(ws + OFF_WB + (size_t)b * SZ_WB);
        for (int k = 0; k < 12; ++k) {
            const WPrep& wp = g_wps[k];
            prep_pad_kernel<<<g1((size_t)wp.N * wp.Kp), 256, 0, stream>>>(
                (const float*)d_in[ib + wp.rel], WB + wp.off, wp.N, wp.K, wp.Kp);
        }
    }
    // inputs -> bf16 [512,160] (staged in the H1 region; consumed before H1 is used)
    __bf16* inputs_bf = (__bf16*)(ws + OFF_H1);
    inputs_bf_kernel<<<g1((size_t)NSMP * 160), 256, 0, stream>>>(inputs, inputs_bf);

    // --- init layer: per-position masked GEMM, writes X position-major ---
    // batch z = pos: C[n, h] -> X[(n*128+z)*256 + h]; bias[h*128+z]
    gemm_go(stream, inputs_bf, 160, 0,
            (const __bf16*)(ws + OFF_WI), 160, (long long)256 * 160,
            (const float*)d_in[2], 128, 1,
            (float*)(ws + OFF_X), 32768, 256,
            NSMP, 256, 160, 256, 0, 128);

    // --- attention blocks ---
    for (int b = 0; b < 4; ++b) run_block(stream, d_in, ws, b);

    // --- final masked GEMM: out[512,2064] = relu(X) @ Wf^T + bf ---
    relu_abf_kernel<<<g1((size_t)NSMP * 32768), 256, 0, stream>>>(
        (const float*)(ws + OFF_X), (__bf16*)(ws + OFF_E));
    gemm_go(stream, (const __bf16*)(ws + OFF_E), 32768, 0,
            (const __bf16*)(ws + OFF_WF), 32768, 0,
            (const float*)d_in[4], 1, 0,
            (float*)d_out, 2064, 0,
            NSMP, 2064, 32768, 2064, 0, 1);
}
